// GatedGCNNet_21071109554392
// MI455X (gfx1250) — matrix operand order repaired and checked by reading, not compiled, not run on hardware
//
#include <hip/hip_runtime.h>

// ---------------------------------------------------------------------------
// GatedGCN layer for MI455X (gfx1250, wave32, WMMA).
//   x   = X @ W1                 (WMMA bf16, fp32 accum)
//   xv  = x @ v                  (WMMA; exploits gather/GEMM commutation)
//   accum[b,tgt,:] += xv[b,src,:] * ew[e] * w2[:]   (fp32 atomics, L2-resident)
//   out = x @ u + accum/max(cnt,1); LayerNorm over (B,C) per node; x + relu
//
// Weights live in LDS in *fragment-major* bf16 layout so a B fragment is one
// 32B contiguous per-lane load (2x ds_load_b128) instead of 16x ds_load_u16.
// ---------------------------------------------------------------------------

typedef __attribute__((ext_vector_type(16))) __bf16 v16bf;
typedef __attribute__((ext_vector_type(8)))  float  v8f;

#define C128 128
#define NBATCH 8

__device__ __forceinline__ v8f wmma_bf16(v16bf a, v16bf b, v8f c) {
    // 8 args: (neg_a, A, neg_b, B, c_mod, C, reuse_a, reuse_b)
    return __builtin_amdgcn_wmma_f32_16x16x32_bf16(false, a, false, b, (short)0, c,
                                                   false, false);
}

// A fragment (16x32 bf16) loaded straight from row-major fp32 global memory.
// ISA layout: lanes 0-15 hold row M=lane, K = {k0..k0+7, k0+16..k0+23};
//             lanes 16-31 hold row M=lane-16, K = {k0+8..k0+15, k0+24..k0+31}.
__device__ __forceinline__ v16bf load_a_frag_global(const float* __restrict__ rows,
                                                    int k0, int lane) {
    int m = lane & 15, half = lane >> 4;
    const float* r = rows + m * C128;
    const float4* p0 = (const float4*)(r + k0 + half * 8);
    const float4* p1 = (const float4*)(r + k0 + 16 + half * 8);
    float4 f0 = p0[0], f1 = p0[1], f2 = p1[0], f3 = p1[1];
    v16bf a;
    a[0]  = (__bf16)f0.x; a[1]  = (__bf16)f0.y; a[2]  = (__bf16)f0.z; a[3]  = (__bf16)f0.w;
    a[4]  = (__bf16)f1.x; a[5]  = (__bf16)f1.y; a[6]  = (__bf16)f1.z; a[7]  = (__bf16)f1.w;
    a[8]  = (__bf16)f2.x; a[9]  = (__bf16)f2.y; a[10] = (__bf16)f2.z; a[11] = (__bf16)f2.w;
    a[12] = (__bf16)f3.x; a[13] = (__bf16)f3.y; a[14] = (__bf16)f3.z; a[15] = (__bf16)f3.w;
    return a;
}

// Stage a 128x128 fp32 weight matrix into LDS as bf16 in fragment-major order.
// Fragment f = kt*8+nt (kt: K/32 step, nt: N/16 tile). Within a fragment each
// lane's 16 elements are contiguous: ws[((f*32+lane)<<4)+e] where the ISA
// B-layout wants  b[e] = W[(kt*32 + (lane>>4)*8 + (e&7) + (e>>3)*16)*128 + nt*16 + (lane&15)].
// Global reads stay coalesced; the scatter lands on cheap LDS stores.
__device__ __forceinline__ void stage_weight_swizzled(__bf16* ws,
                                                      const float* __restrict__ W,
                                                      int tid) {
    for (int j = tid; j < C128 * C128; j += 256) {
        int k = j >> 7, n = j & 127;
        int kt = k >> 5, e = (k & 7) + (((k >> 4) & 1) << 3), halfk = (k >> 3) & 1;
        int lane = halfk * 16 + (n & 15);
        int f = kt * 8 + (n >> 4);
        ws[((f * 32 + lane) << 4) + e] = (__bf16)W[j];
    }
}

// One 32B contiguous per-lane load -> 2x ds_load_b128, bank-conflict free.
__device__ __forceinline__ v16bf load_b_frag_sw(const __bf16* ws, int kt, int nt,
                                                int lane) {
    return *(const v16bf*)(ws + (((kt * 8 + nt) * 32 + lane) << 4));
}

// C/D layout: VGPR r, lanes 0-15 -> (M=r, N=lane); lanes 16-31 -> (M=r+8).
__device__ __forceinline__ void store_acc(float* __restrict__ o, v8f c, int lane) {
    int n = lane & 15, half = lane >> 4;
#pragma unroll
    for (int r = 0; r < 8; ++r) o[(size_t)(half * 8 + r) * C128 + n] = c[r];
}

// ---------------------------------------------------------------------------
// out[r,:] = in[r,:] @ W   for `tiles` row-tiles of 16; W is 128x128 fp32.
// 8 waves / block, one 16x128 tile per wave, W staged once in LDS (bf16, 32KB).
// ---------------------------------------------------------------------------
__global__ __launch_bounds__(256) void gemm128_kernel(const float* __restrict__ in,
                                                      const float* __restrict__ W,
                                                      float* __restrict__ out,
                                                      int tiles) {
    __shared__ __bf16 w_s[C128 * C128];
    int tid = threadIdx.x;
    stage_weight_swizzled(w_s, W, tid);
    __syncthreads();

    int lane = tid & 31, wave = tid >> 5;
    for (int tile = blockIdx.x * 8 + wave; tile < tiles; tile += gridDim.x * 8) {
        const float* arow = in + (size_t)tile * 16 * C128;
        v16bf af[4];
#pragma unroll
        for (int kt = 0; kt < 4; ++kt) af[kt] = load_a_frag_global(arow, kt * 32, lane);

        v8f acc[8];
#pragma unroll
        for (int nt = 0; nt < 8; ++nt)
#pragma unroll
            for (int i = 0; i < 8; ++i) acc[nt][i] = 0.0f;

#pragma unroll
        for (int kt = 0; kt < 4; ++kt)
#pragma unroll
            for (int nt = 0; nt < 8; ++nt) {
                v16bf bf = load_b_frag_sw(w_s, kt, nt, lane);
                acc[nt] = wmma_bf16(af[kt], bf, acc[nt]);
            }

        float* orow = out + (size_t)tile * 16 * C128;
#pragma unroll
        for (int nt = 0; nt < 8; ++nt) store_acc(orow + nt * 16, acc[nt], lane);
    }
}

// ---------------------------------------------------------------------------
// One block per edge; wave b handles batch b (8 waves).
// accum[b,tgt,:] += xv[b,src,:] * ew[e] * w2[:]; counts[tgt] += 1 (once).
// xv and accum are each 41MB -> resident in the 192MB L2.
// ---------------------------------------------------------------------------
__global__ __launch_bounds__(256) void edge_kernel(const long long* __restrict__ ei,
                                                   const float* __restrict__ ew,
                                                   const float* __restrict__ w2,
                                                   const float* __restrict__ xv,
                                                   float* __restrict__ accum,
                                                   float* __restrict__ counts,
                                                   int E, int N) {
    int e = blockIdx.x;
    if (e >= E) return;
    int tid = threadIdx.x, b = tid >> 5, lane = tid & 31;
    int src = (int)ei[e];
    int tgt = (int)ei[(size_t)E + e];
    float w = ew[e];
    if (b == 0 && lane == 0) atomicAdd(&counts[tgt], 1.0f);

    const float4* xr  = (const float4*)(xv + ((size_t)b * N + src) * C128);
    const float4* w2r = (const float4*)w2;
    float4 xvv = xr[lane];
    float4 ww  = w2r[lane];
    float* ar = accum + ((size_t)b * N + tgt) * C128 + lane * 4;
    atomicAdd(ar + 0, xvv.x * w * ww.x);
    atomicAdd(ar + 1, xvv.y * w * ww.y);
    atomicAdd(ar + 2, xvv.z * w * ww.z);
    atomicAdd(ar + 3, xvv.w * w * ww.w);
}

// ---------------------------------------------------------------------------
// Finale, one block per 16-node tile: out = x@u (WMMA) + aggr; LayerNorm over
// (B,C) per node; write x + relu(norm). 8 waves <-> 8 batches.
// ---------------------------------------------------------------------------
__global__ __launch_bounds__(256) void final_kernel(const float* __restrict__ x,
                                                    const float* __restrict__ u,
                                                    const float* __restrict__ accum,
                                                    const float* __restrict__ counts,
                                                    float* __restrict__ out, int N) {
    __shared__ __bf16 u_s[C128 * C128];
    __shared__ float inv_s[16], sum_s[16], ssq_s[16];

    int tid = threadIdx.x;
    stage_weight_swizzled(u_s, u, tid);
    int n0g = blockIdx.x * 16;
    if (tid < 16) {
        inv_s[tid] = 1.0f / fmaxf(counts[n0g + tid], 1.0f);
        sum_s[tid] = 0.0f;
        ssq_s[tid] = 0.0f;
    }
    __syncthreads();

    int lane = tid & 31, b = tid >> 5;
    int half = lane >> 4, nn = lane & 15;
    const float* xrows = x + ((size_t)b * N + n0g) * C128;

    v16bf af[4];
#pragma unroll
    for (int kt = 0; kt < 4; ++kt) af[kt] = load_a_frag_global(xrows, kt * 32, lane);

    v8f acc[8];
#pragma unroll
    for (int nt = 0; nt < 8; ++nt)
#pragma unroll
        for (int i = 0; i < 8; ++i) acc[nt][i] = 0.0f;

#pragma unroll
    for (int kt = 0; kt < 4; ++kt)
#pragma unroll
        for (int nt = 0; nt < 8; ++nt) {
            v16bf bf = load_b_frag_sw(u_s, kt, nt, lane);
            acc[nt] = wmma_bf16(af[kt], bf, acc[nt]);
        }

    // add mean-aggregated messages
    const float* arows = accum + ((size_t)b * N + n0g) * C128;
#pragma unroll
    for (int nt = 0; nt < 8; ++nt)
#pragma unroll
        for (int r = 0; r < 8; ++r) {
            int node = half * 8 + r, col = nt * 16 + nn;
            acc[nt][r] += arows[(size_t)node * C128 + col] * inv_s[node];
        }

    // per-node stats over (batch, channel): shfl-reduce the 16 lanes of each
    // half (xor masks 1..8 stay inside the half), then LDS atomics across waves.
#pragma unroll
    for (int nt = 0; nt < 8; ++nt)
#pragma unroll
        for (int r = 0; r < 8; ++r) {
            float s = acc[nt][r], q = s * s;
#pragma unroll
            for (int off = 1; off < 16; off <<= 1) {
                s += __shfl_xor(s, off);
                q += __shfl_xor(q, off);
            }
            if (nn == 0) {
                int node = half * 8 + r;
                atomicAdd(&sum_s[node], s);
                atomicAdd(&ssq_s[node], q);
            }
        }
    __syncthreads();

    if (tid < 16) {
        const float inv_n = 1.0f / (float)(NBATCH * C128);
        float mean = sum_s[tid] * inv_n;
        float var  = ssq_s[tid] * inv_n - mean * mean;
        sum_s[tid] = mean;
        ssq_s[tid] = rsqrtf(var + 1e-5f);
    }
    __syncthreads();

    float* orows = out + ((size_t)b * N + n0g) * C128;
#pragma unroll
    for (int nt = 0; nt < 8; ++nt)
#pragma unroll
        for (int r = 0; r < 8; ++r) {
            int node = half * 8 + r, col = nt * 16 + nn;
            float o = (acc[nt][r] - sum_s[node]) * ssq_s[node];
            o = fmaxf(o, 0.0f);
            orows[(size_t)node * C128 + col] = xrows[(size_t)node * C128 + col] + o;
        }
}

__global__ void zero_kernel(float* __restrict__ p, size_t n) {
    size_t i = (size_t)blockIdx.x * blockDim.x + threadIdx.x;
    size_t stride = (size_t)gridDim.x * blockDim.x;
    for (; i < n; i += stride) p[i] = 0.0f;
}

// ---------------------------------------------------------------------------
extern "C" void kernel_launch(void* const* d_in, const int* in_sizes, int n_in,
                              void* d_out, int out_size, void* d_ws, size_t ws_size,
                              hipStream_t stream) {
    const float*     X  = (const float*)d_in[0];      // (B, N, 128) fp32
    const long long* ei = (const long long*)d_in[1];  // (2, E) int64
    const float*     ew = (const float*)d_in[2];      // (E,) fp32
    const float*     W1 = (const float*)d_in[3];      // (128, 128)
    const float*     w2 = (const float*)d_in[4];      // (1, 128)
    const float*     u  = (const float*)d_in[5];      // (128, 128)
    const float*     v  = (const float*)d_in[6];      // (128, 128)
    float* out = (float*)d_out;

    const int B = NBATCH;
    const int N = in_sizes[0] / (B * C128);  // 10000
    const int E = in_sizes[2];               // 160000

    const size_t node_elems = (size_t)B * N * C128;
    float* xbuf   = (float*)d_ws;            // x  = X@W1
    float* xvbuf  = xbuf + node_elems;       // xv = x@v
    float* accum  = xvbuf + node_elems;      // message accumulator
    float* counts = accum + node_elems;      // degree counts (contiguous after accum)

    // 1) zero accumulator + counts (harness poisons ws; must re-zero each call)
    zero_kernel<<<2048, 256, 0, stream>>>(accum, node_elems + (size_t)N);

    // 2) x = X @ W1 ; 3) xv = x @ v
    int tiles = (B * N) / 16;  // 5000
    int gblocks = (tiles + 7) / 8;
    gemm128_kernel<<<gblocks, 256, 0, stream>>>(X, W1, xbuf, tiles);
    gemm128_kernel<<<gblocks, 256, 0, stream>>>(xbuf, v, xvbuf, tiles);

    // 4) edge scatter (mean-aggregation numerators + counts)
    edge_kernel<<<E, 256, 0, stream>>>(ei, ew, w2, xvbuf, accum, counts, E, N);

    // 5) x@u + aggr, LayerNorm over (B,C), relu, residual
    final_kernel<<<N / 16, 256, 0, stream>>>(xbuf, u, accum, counts, out, N);
}
